// Attention_14989435863424
// MI455X (gfx1250) — compile-verified, matrix operand order checked by reading, not run on hardware
//
#include <hip/hip_runtime.h>
#include <hip/hip_bf16.h>

// ---- problem constants (from reference) ----
#define B_  8
#define LQ_ 2048
#define LK_ 2048
#define D_  512
#define QT  16          // query rows per workgroup
#define KB  128         // keys per block iteration (16 per wave)
#define WAVES 8
#define NTHREADS 256
#define NEGV (-1.0e9f)

// LDS strides (bf16 elements), padded (+8 elems = 16B) to rotate banks
#define QS_STRIDE (D_ + 8)      // Q tile rows
#define KS_STRIDE (D_ + 8)      // K block rows (key-major)
#define VT_STRIDE (KB + 8)      // V block rows (d-major, transposed)
#define PS_STRIDE (KB + 8)      // P tile rows

typedef __attribute__((ext_vector_type(16))) __bf16 v16bf;
typedef __attribute__((ext_vector_type(8)))  __bf16 v8bf;
typedef __attribute__((ext_vector_type(4)))  __bf16 v4bf;
typedef __attribute__((ext_vector_type(8)))  float  v8f;

// dynamic LDS layout size in bytes (host-visible constant)
enum { SMEM_BYTES = (QT*QS_STRIDE + D_*VT_STRIDE + QT*PS_STRIDE) * 2
                    + (2*QT*WAVES + 2*QT) * 4 };

__device__ __forceinline__ v16bf cat16(v8bf lo, v8bf hi) {
  return __builtin_shufflevector(lo, hi, 0,1,2,3,4,5,6,7,8,9,10,11,12,13,14,15);
}

__global__ __launch_bounds__(NTHREADS, 1)
void fa_fwd_kernel(const float* __restrict__ Q, const float* __restrict__ K,
                   const float* __restrict__ V, const float* __restrict__ ratio_p,
                   const float* __restrict__ scale_p,
                   const unsigned char* __restrict__ mask_p,
                   float* __restrict__ Out)
{
  extern __shared__ char smem_raw[];
  __bf16* qs = (__bf16*)smem_raw;                       // [QT][QS_STRIDE]
  __bf16* kv = qs + QT * QS_STRIDE;                     // shared K ([KB][KS_STRIDE]) / V^T ([D_][VT_STRIDE]) buffer
  __bf16* ps = kv + (size_t)D_ * VT_STRIDE;             // [QT][PS_STRIDE]
  float*  redA  = (float*)(ps + QT * PS_STRIDE);        // [QT][WAVES] row-max partials
  float*  redB  = redA + QT * WAVES;                    // [QT][WAVES] row-sum partials
  float*  m_run = redB + QT * WAVES;                    // [QT] running max
  float*  l_run = m_run + QT;                           // [QT] running denom

  const int tid  = threadIdx.x;
  const int w    = tid >> 5;      // wave id 0..7
  const int lane = tid & 31;
  const int h    = lane >> 4;     // lane half (WMMA layout)
  const int ln   = lane & 15;

  const int b  = blockIdx.y;
  const int q0 = blockIdx.x * QT;

  const float* Qb = Q + ((size_t)b * LQ_ + q0) * D_;
  const float* Kb = K + (size_t)b * LK_ * D_;
  const float* Vb = V + (size_t)b * LK_ * D_;
  const unsigned char* Mb = mask_p + (size_t)b * LK_;
  const float ratio = ratio_p[b];
  const float sr    = scale_p[0] * ratio;   // applied to unmasked scores
  const float negr  = NEGV * ratio;         // masked score after ratio

  // ---- stage Q tile f32 -> bf16 into LDS (b128 reads, b64 packed stores) ----
  for (int i = tid; i < QT * D_ / 4; i += NTHREADS) {
    const int r = (i << 2) / D_;
    const int c = (i << 2) & (D_ - 1);
    const float4 f = *(const float4*)(Qb + (size_t)r * D_ + c);
    v4bf pk = { (__bf16)f.x, (__bf16)f.y, (__bf16)f.z, (__bf16)f.w };
    *(v4bf*)(qs + r * QS_STRIDE + c) = pk;
  }
  if (tid < QT) { m_run[tid] = -3.0e38f; l_run[tid] = 0.0f; }

  v8f oacc[4];
  {
    v8f z = {};
    #pragma unroll
    for (int t = 0; t < 4; ++t) oacc[t] = z;
  }

  for (int j0 = 0; j0 < LK_; j0 += KB) {
    __syncthreads();   // Q staged / previous P*V reads of kv complete

    // ---- stage K block f32 -> bf16 into kv as [key][d] ----
    for (int i = tid; i < KB * D_ / 4; i += NTHREADS) {
      const int key = (i << 2) / D_;
      const int c   = (i << 2) & (D_ - 1);
      const float4 f = *(const float4*)(Kb + (size_t)(j0 + key) * D_ + c);
      v4bf pk = { (__bf16)f.x, (__bf16)f.y, (__bf16)f.z, (__bf16)f.w };
      *(v4bf*)(kv + key * KS_STRIDE + c) = pk;
    }
    if (j0 + KB < LK_) {   // pull next block toward L2 (global_prefetch_b8)
      __builtin_prefetch(Kb + (size_t)(j0 + KB) * D_ + (size_t)tid * 64, 0, 1);
      __builtin_prefetch(Vb + (size_t)(j0 + KB) * D_ + (size_t)tid * 64, 0, 1);
    }
    __syncthreads();

    // ---- S = Q * K^T  (this wave: 16 queries x 16 keys, K-dim 512) ----
    v8f s = {};
    const __bf16* qrow = qs + ln * QS_STRIDE + h * 8;                 // A: row = ln
    const __bf16* krow = kv + (w * 16 + ln) * KS_STRIDE + h * 8;      // B: col(key) = ln
    #pragma unroll 4
    for (int dc = 0; dc < D_ / 32; ++dc) {
      const int base = dc * 32;
      const v8bf alo = *(const v8bf*)(qrow + base);        // K-run [h*8, h*8+8)
      const v8bf ahi = *(const v8bf*)(qrow + base + 16);   // K-run [16+h*8, ..)
      const v8bf blo = *(const v8bf*)(krow + base);
      const v8bf bhi = *(const v8bf*)(krow + base + 16);
      s = __builtin_amdgcn_wmma_f32_16x16x32_bf16(false, cat16(alo, ahi),
                                                  false, cat16(blo, bhi),
                                                  (short)0, s, false, false);
    }

    // ---- scale, mask-fill, ratio ----
    const bool mk = Mb[j0 + w * 16 + ln] != 0;
    #pragma unroll
    for (int i = 0; i < 8; ++i) s[i] = mk ? negr : s[i] * sr;

    // ---- per-row max within wave (N striped across 16-lane halves) ----
    #pragma unroll
    for (int i = 0; i < 8; ++i) {
      float m = s[i];
      m = fmaxf(m, __shfl_xor(m, 1));
      m = fmaxf(m, __shfl_xor(m, 2));
      m = fmaxf(m, __shfl_xor(m, 4));
      m = fmaxf(m, __shfl_xor(m, 8));
      if (ln == 0) redA[(i + 8 * h) * WAVES + w] = m;
    }
    __syncthreads();

    // ---- block max + online-softmax update (all waves identical result) ----
    float mnew[8], alpha[8];
    #pragma unroll
    for (int i = 0; i < 8; ++i) {
      const int row = i + 8 * h;
      float bm = redA[row * WAVES];
      #pragma unroll
      for (int wv = 1; wv < WAVES; ++wv) bm = fmaxf(bm, redA[row * WAVES + wv]);
      const float mo = m_run[row];
      const float mn = fmaxf(mo, bm);
      mnew[i]  = mn;
      alpha[i] = __expf(mo - mn);
    }
    __syncthreads();
    if (w == 0 && ln == 0) {
      #pragma unroll
      for (int i = 0; i < 8; ++i) m_run[i + 8 * h] = mnew[i];
    }

    // ---- p = exp(s - mnew); partial sums; P tile -> LDS bf16 ----
    #pragma unroll
    for (int i = 0; i < 8; ++i) {
      float p = __expf(s[i] - mnew[i]);
      ps[(i + 8 * h) * PS_STRIDE + w * 16 + ln] = (__bf16)p;
      p += __shfl_xor(p, 1);
      p += __shfl_xor(p, 2);
      p += __shfl_xor(p, 4);
      p += __shfl_xor(p, 8);
      if (ln == 0) redB[(i + 8 * h) * WAVES + w] = p;
    }

    // ---- stage V block transposed into kv as [d][key] (bf16) ----
    for (int i = tid; i < KB * D_ / 4; i += NTHREADS) {
      const int key = (i << 2) / D_;
      const int c   = (i << 2) & (D_ - 1);
      const float4 f = *(const float4*)(Vb + (size_t)(j0 + key) * D_ + c);
      kv[(c + 0) * VT_STRIDE + key] = (__bf16)f.x;
      kv[(c + 1) * VT_STRIDE + key] = (__bf16)f.y;
      kv[(c + 2) * VT_STRIDE + key] = (__bf16)f.z;
      kv[(c + 3) * VT_STRIDE + key] = (__bf16)f.w;
    }
    __syncthreads();

    // ---- denominator update (wave 0 only; others never read l_run here) ----
    if (w == 0 && ln == 0) {
      #pragma unroll
      for (int i = 0; i < 8; ++i) {
        const int row = i + 8 * h;
        float bs = redB[row * WAVES];
        #pragma unroll
        for (int wv = 1; wv < WAVES; ++wv) bs += redB[row * WAVES + wv];
        l_run[row] = alpha[i] * l_run[row] + bs;
      }
    }

    // ---- rescale O accumulator by alpha ----
    #pragma unroll
    for (int t = 0; t < 4; ++t)
      #pragma unroll
      for (int i = 0; i < 8; ++i) oacc[t][i] *= alpha[i];

    // ---- O += P * V   (this wave: 16 queries x 64 d-cols, K-dim 128) ----
    // kc outer / t inner: each P A-fragment loaded once, reused by 4 WMMAs.
    const __bf16* prow = ps + ln * PS_STRIDE + h * 8;                 // A: row = ln
    const __bf16* vcol = kv + (w * 64 + ln) * VT_STRIDE + h * 8;      // B: col(d) = ln
    #pragma unroll
    for (int kc = 0; kc < 4; ++kc) {
      const int kb2 = kc * 32;
      const v8bf alo = *(const v8bf*)(prow + kb2);
      const v8bf ahi = *(const v8bf*)(prow + kb2 + 16);
      const v16bf a = cat16(alo, ahi);
      #pragma unroll
      for (int t = 0; t < 4; ++t) {
        const __bf16* vrow = vcol + (size_t)(t * 16) * VT_STRIDE;
        const v8bf blo = *(const v8bf*)(vrow + kb2);
        const v8bf bhi = *(const v8bf*)(vrow + kb2 + 16);
        oacc[t] = __builtin_amdgcn_wmma_f32_16x16x32_bf16(false, a,
                                                          false, cat16(blo, bhi),
                                                          (short)0, oacc[t], false, false);
      }
    }
  }

  __syncthreads();  // final l_run writes visible

  // ---- normalize and write out (fp32) ----
  #pragma unroll
  for (int i = 0; i < 8; ++i) {
    const int row = i + 8 * h;
    const float inv = 1.0f / l_run[row];
    float* orow = Out + ((size_t)b * LQ_ + q0 + row) * D_ + w * 64 + ln;
    #pragma unroll
    for (int t = 0; t < 4; ++t) orow[t * 16] = oacc[t][i] * inv;
  }
}

extern "C" void kernel_launch(void* const* d_in, const int* in_sizes, int n_in,
                              void* d_out, int out_size, void* d_ws, size_t ws_size,
                              hipStream_t stream) {
  (void)in_sizes; (void)n_in; (void)out_size; (void)d_ws; (void)ws_size;
  const float* q     = (const float*)d_in[0];
  const float* k     = (const float*)d_in[1];
  const float* v     = (const float*)d_in[2];
  const float* ratio = (const float*)d_in[3];
  const float* scale = (const float*)d_in[4];
  const unsigned char* mask = (const unsigned char*)d_in[5];  // jax bool_ = 1 byte
  float* out = (float*)d_out;

  // dynamic LDS ~158 KB > default cap: opt in once (idempotent, capture-safe)
  static bool attr_set = false;
  if (!attr_set) {
    hipFuncSetAttribute((const void*)fa_fwd_kernel,
                        hipFuncAttributeMaxDynamicSharedMemorySize, SMEM_BYTES);
    attr_set = true;
  }

  dim3 grid(LQ_ / QT, B_);
  fa_fwd_kernel<<<grid, NTHREADS, SMEM_BYTES, stream>>>(q, k, v, ratio, scale, mask, out);
}
